// MultiScaleRetention_463856468165
// MI455X (gfx1250) — matrix operand order repaired and checked by reading, not compiled, not run on hardware
//
#include <hip/hip_runtime.h>
#include <hip/hip_bf16.h>

// ---------------------------------------------------------------------------
// MultiScaleRetention on MI455X (gfx1250), wave32 + WMMA f32_16x16x32_f16.
// ~77 GFLOP total; fused retention avoids the 268MB score tensor.
// Round 2: K/V^T slab staging now uses GLOBAL_LOAD_ASYNC_TO_LDS_B128
// (ASYNCcnt-tracked, no VGPR round-trip) + global_prefetch_b8 in GEMMs.
// ---------------------------------------------------------------------------

#define HID    1024
#define NHEADS 8
#define HEAD   128
#define BATCH  2
#define SEQ    2048
#define GN_EPS 1e-5f

typedef __attribute__((ext_vector_type(16))) _Float16 v16h;
typedef __attribute__((ext_vector_type(8)))  _Float16 v8h;
typedef __attribute__((ext_vector_type(8)))  float    v8f;
typedef __attribute__((ext_vector_type(4)))  float    v4f;

// ---------------- WMMA helpers (CDNA5 operand layouts, ISA 7.12.2) ---------

static __device__ __forceinline__ v8f wmma32(v16h a, v16h b, v8f c) {
  // D = A(16x32 f16) * B(32x16 f16) + C(16x16 f32)
  return __builtin_amdgcn_wmma_f32_16x16x32_f16(
      /*neg_a=*/false, a, /*neg_b=*/false, b,
      /*c_mod=*/(short)0, c, /*reuse_a=*/false, /*reuse_b=*/false);
}

// A-matrix 16x32 f16, row-major source with leading dim `lda`.
// Lane L<16 : M=L, K in {0..7, 16..23}; lane L>=16 : M=L-16, K in {8..15, 24..31}
static __device__ __forceinline__ v16h load_a_frag(const _Float16* __restrict__ base,
                                                   int lda, int row0, int k0) {
  const int lane = threadIdx.x & 31;
  const int m  = lane & 15;
  const int kh = (lane >> 4) * 8;
  const _Float16* p = base + (size_t)(row0 + m) * lda + k0 + kh;
  v8h lo = *(const v8h*)(p);        // K k0+kh .. +7
  v8h hi = *(const v8h*)(p + 16);   // K k0+kh+16 .. +23
  v16h f;
#pragma unroll
  for (int i = 0; i < 8; ++i) { f[i] = lo[i]; f[8 + i] = hi[i]; }
  return f;
}

// B-matrix 32x16 f16 read from a PRE-TRANSPOSED operand Bt[n][k] (ldb = K dim).
// Lane L<16 : N=L, K in {0..7, 16..23} ; lane L>=16 : N=L-16, K in {8..15, 24..31}.
static __device__ __forceinline__ v16h load_bt_frag(const _Float16* __restrict__ bt,
                                                    int ldb, int n0, int k0) {
  const int lane = threadIdx.x & 31;
  const int n  = lane & 15;
  const int kh = (lane >> 4) * 8;
  const _Float16* p = bt + (size_t)(n0 + n) * ldb + k0 + kh;
  v8h lo = *(const v8h*)(p);
  v8h hi = *(const v8h*)(p + 16);
  v16h f;
#pragma unroll
  for (int i = 0; i < 8; ++i) { f[i] = lo[i]; f[8 + i] = hi[i]; }
  return f;
}

// C/D layout: VGPR r, lanes0-15 -> M=r, lanes16-31 -> M=r+8 ; N = lane&15.

// ---------------- async global -> LDS (CDNA5, ASYNCcnt) --------------------
// Moves 16B per lane directly into LDS, no VGPR staging. LDS byte address is
// the low 32 bits of the generic shared pointer.
static __device__ __forceinline__ void async_ld_b128(_Float16* lds_dst,
                                                     const _Float16* gsrc) {
  unsigned       lds = (unsigned)(uintptr_t)lds_dst;
  unsigned long long ga = (unsigned long long)(uintptr_t)gsrc;
  asm volatile("global_load_async_to_lds_b128 %0, %1, off"
               :: "v"(lds), "v"(ga) : "memory");
}
static __device__ __forceinline__ void wait_asynccnt0() {
  asm volatile("s_wait_asynccnt 0x0" ::: "memory");
}

// ---------------- gammas ---------------------------------------------------

static __device__ __forceinline__ float gamma_h(int h) {
  const float l0 = -3.4657359027997265f;   // log(1/32)
  const float l1 = -6.2383246250395075f;   // log(1/512)
  float lg = l0 + (l1 - l0) * ((float)h * (1.0f / 7.0f));
  return 1.0f - __expf(lg);
}

// ---------------- elementwise convert --------------------------------------

__global__ void f32_to_f16_kernel(const float* __restrict__ in,
                                  _Float16* __restrict__ out, int n) {
  int i = blockIdx.x * 256 + threadIdx.x;
  if (i < n) out[i] = (_Float16)in[i];
}

// ---------------- tiled transpose+convert: f32 [R][C] -> f16 [C][R] --------

__global__ __launch_bounds__(256)
void transpose_to_f16_kernel(const float* __restrict__ in,
                             _Float16* __restrict__ out, int R, int C) {
  __shared__ float tile[32][33];
  const int mat = blockIdx.z;
  const float*  src = in  + (size_t)mat * R * C;
  _Float16*     dst = out + (size_t)mat * R * C;
  const int c0 = blockIdx.x * 32, r0 = blockIdx.y * 32;
  const int tx = threadIdx.x & 31, ty = threadIdx.x >> 5;
#pragma unroll
  for (int i = 0; i < 32; i += 8)
    tile[ty + i][tx] = src[(size_t)(r0 + ty + i) * C + c0 + tx];
  __syncthreads();
#pragma unroll
  for (int i = 0; i < 32; i += 8)
    dst[(size_t)(c0 + ty + i) * R + r0 + tx] = (_Float16)tile[tx][ty + i];
}

// ---------------- QKV projection + xpos rotary epilogue --------------------
// grid: (SEQ/128, BATCH*NHEADS, 3)  block: 256 (8 waves, 16 rows each)
__global__ __launch_bounds__(256)
void proj_xpos_kernel(const _Float16* __restrict__ Xh,
                      const _Float16* __restrict__ Wqt,
                      const _Float16* __restrict__ Wkt,
                      const _Float16* __restrict__ Wvt,
                      _Float16* __restrict__ Qh,
                      _Float16* __restrict__ Kh,
                      _Float16* __restrict__ Vt) {
  const int sel  = blockIdx.z;                 // 0=Q 1=K 2=V
  const int b    = blockIdx.y / NHEADS;
  const int h    = blockIdx.y % NHEADS;
  const int wave = threadIdx.x >> 5;
  const int lane = threadIdx.x & 31;
  const int s0   = blockIdx.x * 128 + wave * 16;

  const _Float16* Wt = (sel == 0) ? Wqt : (sel == 1) ? Wkt : Wvt;
  const _Float16* A  = Xh + (size_t)b * SEQ * HID;
  const _Float16* Bt = Wt + (size_t)h * HEAD * HID;

  v8f acc[8];
#pragma unroll
  for (int j = 0; j < 8; ++j) acc[j] = (v8f)0.0f;

  for (int k = 0; k < HID; k += 32) {          // 32 K-steps
    v16h a = load_a_frag(A, HID, s0, k);
    if (k + 32 < HID)                          // speculative prefetch of next A slab
      __builtin_prefetch(A + (size_t)(s0 + (lane & 15)) * HID + k + 32, 0, 3);
#pragma unroll
    for (int j = 0; j < 8; ++j) {
      v16h bf = load_bt_frag(Bt, HID, j * 16, k);
      acc[j] = wmma32(a, bf, acc[j]);
    }
  }

  const int mh = (lane >> 4) * 8;
  const int n  = lane & 15;

  if (sel == 2) {                              // V: store TRANSPOSED [d][s]
    _Float16* Vb = Vt + (size_t)(b * NHEADS + h) * HEAD * SEQ;
#pragma unroll
    for (int j = 0; j < 8; ++j)
#pragma unroll
      for (int r = 0; r < 8; ++r) {
        int d = j * 16 + n;
        int s = s0 + r + mh;
        Vb[(size_t)d * SEQ + s] = (_Float16)acc[j][r];
      }
    return;
  }

  // xpos rotary: y[2i] = x[2i]*c*sc - x[2i+1]*s*sc ; y[2i+1] = x[2i+1]*c*sc + x[2i]*s*sc
  _Float16* O = ((sel == 0) ? Qh : Kh) + (size_t)(b * NHEADS + h) * SEQ * HEAD;
  const float sgn = (sel == 0) ? 1.0f : -1.0f;           // K uses 1/scale
  const float L2_10000 = 13.287712379549449f;
#pragma unroll
  for (int j = 0; j < 8; ++j) {
    const int d   = j * 16 + n;
    const int i2  = d >> 1;
    const bool odd = (d & 1);
    const float invf = exp2f(-((float)i2) * (L2_10000 / 64.0f));       // 10000^(-i/64)
    const float lsv  = log2f(((float)(2 * i2) + 0.4f * 128.0f) * (1.0f / (1.4f * 128.0f)));
#pragma unroll
    for (int r = 0; r < 8; ++r) {
      const int   s   = s0 + r + mh;
      const float pos = (float)s;
      const float sc  = exp2f(sgn * lsv * pos * (1.0f / 512.0f));
      const float ang = pos * invf;
      const float cp  = __cosf(ang) * sc;
      const float sp  = __sinf(ang) * sc;
      const float v   = acc[j][r];
      const float pv  = __shfl_xor(v, 1, 32);            // partner column d^1
      const float out = odd ? (cp * v + sp * pv) : (cp * v - sp * pv);
      O[(size_t)s * HEAD + d] = (_Float16)out;
    }
  }
}

// ---------------- generic 128x128-tile GEMM: C = A(MxK) * B, Bt[N][K] ------
// EPI 0: f32 store    EPI 1: SiLU -> f16 store
template <int EPI>
__global__ __launch_bounds__(256)
void gemm_kernel(const _Float16* __restrict__ A, const _Float16* __restrict__ Bt,
                 void* __restrict__ Cout, int K, int N) {
  const int wave = threadIdx.x >> 5;
  const int lane = threadIdx.x & 31;
  const int row0 = blockIdx.x * 128 + wave * 16;
  const int n0   = blockIdx.y * 128;

  v8f acc[8];
#pragma unroll
  for (int j = 0; j < 8; ++j) acc[j] = (v8f)0.0f;

  for (int k = 0; k < K; k += 32) {
    v16h a = load_a_frag(A, K, row0, k);
    if (k + 32 < K)                            // speculative prefetch of next A slab
      __builtin_prefetch(A + (size_t)(row0 + (lane & 15)) * K + k + 32, 0, 3);
#pragma unroll
    for (int j = 0; j < 8; ++j) {
      v16h bf = load_bt_frag(Bt, K, n0 + j * 16, k);
      acc[j] = wmma32(a, bf, acc[j]);
    }
  }

  const int mh = (lane >> 4) * 8;
  const int n  = lane & 15;
#pragma unroll
  for (int j = 0; j < 8; ++j)
#pragma unroll
    for (int r = 0; r < 8; ++r) {
      const size_t idx = (size_t)(row0 + r + mh) * N + n0 + j * 16 + n;
      const float v = acc[j][r];
      if (EPI == 0) {
        ((float*)Cout)[idx] = v;
      } else {
        const float sg = v / (1.0f + __expf(-v));        // SiLU
        ((_Float16*)Cout)[idx] = (_Float16)sg;
      }
    }
}

// ---------------- fused retention: Y^T += V^T * (decay o QK^T)^T -----------
// grid: (SEQ/128, BATCH*NHEADS)  block: 256. Wave w owns rows s0..s0+15.
__global__ __launch_bounds__(256)
void retention_kernel(const _Float16* __restrict__ Qh,
                      const _Float16* __restrict__ Kh,
                      const _Float16* __restrict__ Vt,
                      float* __restrict__ Y) {
  __shared__ __align__(16) _Float16 bufK[32 * HEAD];    // K slab   [t][d]
  __shared__ __align__(16) _Float16 bufV[HEAD * 32];    // V^T slab [d][t]
  __shared__ __align__(16) _Float16 bufP[8][16 * 32];   // per-wave P [s][t]

  const int b    = blockIdx.y / NHEADS;
  const int h    = blockIdx.y % NHEADS;
  const int wave = threadIdx.x >> 5;
  const int lane = threadIdx.x & 31;
  const int s0   = blockIdx.x * 128 + wave * 16;

  const float l2g = log2f(gamma_h(h));

  const _Float16* Qbase = Qh + ((size_t)(b * NHEADS + h) * SEQ + s0) * HEAD;
  const _Float16* Kbase = Kh + (size_t)(b * NHEADS + h) * SEQ * HEAD;
  const _Float16* Vbase = Vt + (size_t)(b * NHEADS + h) * HEAD * SEQ;

  // Q strip 16x128 lives in registers for the whole t-loop (4 A-fragments).
  v16h qf[4];
#pragma unroll
  for (int kk = 0; kk < 4; ++kk) qf[kk] = load_a_frag(Qbase, HEAD, 0, kk * 32);

  v8f yacc[8];                                          // Y^T: 128(dv) x 16(s)
#pragma unroll
  for (int j = 0; j < 8; ++j) yacc[j] = (v8f)0.0f;

  const int mh   = (lane >> 4) * 8;
  const int n    = lane & 15;
  const int tmax = (blockIdx.x + 1) * 128;              // causal bound for WG

  // per-thread slab-copy coordinates (32B per thread per slab)
  const int tid = threadIdx.x;
  const int kr = tid >> 3, kc = (tid & 7) * 16;
  const int vr = tid >> 1, vc = (tid & 1) * 16;

  for (int t0 = 0; t0 < tmax; t0 += 32) {
    // cooperative slab loads, async straight into LDS (no VGPR staging):
    // 8KB K slab + 8KB V^T slab, 4x b128 per thread, tracked by ASYNCcnt.
    async_ld_b128(&bufK[kr * HEAD + kc],     Kbase + (size_t)(t0 + kr) * HEAD + kc);
    async_ld_b128(&bufK[kr * HEAD + kc + 8], Kbase + (size_t)(t0 + kr) * HEAD + kc + 8);
    async_ld_b128(&bufV[vr * 32 + vc],       Vbase + (size_t)vr * SEQ + t0 + vc);
    async_ld_b128(&bufV[vr * 32 + vc + 8],   Vbase + (size_t)vr * SEQ + t0 + vc + 8);
    wait_asynccnt0();
    __syncthreads();

    if (t0 <= s0 + 15) {                                // wave-uniform guard
      // scores P(16x32) = Q(16x128) * K^T : Bt == bufK rows directly
      v8f p0 = (v8f)0.0f, p1 = (v8f)0.0f;
#pragma unroll
      for (int kk = 0; kk < 4; ++kk) {
        p0 = wmma32(qf[kk], load_bt_frag(bufK, HEAD, 0,  kk * 32), p0);
        p1 = wmma32(qf[kk], load_bt_frag(bufK, HEAD, 16, kk * 32), p1);
      }
      // decay * causal mask, stage as P[s][t] f16 (same-wave DS is in-order)
      _Float16* Pw = &bufP[wave][0];
#pragma unroll
      for (int nb = 0; nb < 2; ++nb) {
        v8f p = nb ? p1 : p0;
#pragma unroll
        for (int r = 0; r < 8; ++r) {
          const int s    = s0 + r + mh;
          const int t    = t0 + nb * 16 + n;
          const int diff = s - t;
          const float w  = (diff >= 0) ? exp2f((float)diff * l2g) : 0.0f;
          Pw[(r + mh) * 32 + nb * 16 + n] = (_Float16)(p[r] * w);
        }
      }
      // Y^T(128x16) += V^T(128x32) * P^T(32x16):
      //   A = bufV rows, Bt[n=s][k=t] = P[s][t] = Pw rows.
      v16h bp = load_bt_frag(Pw, 32, 0, 0);
#pragma unroll
      for (int j = 0; j < 8; ++j) {
        v16h av = load_a_frag(bufV, 32, j * 16, 0);
        yacc[j] = wmma32(av, bp, yacc[j]);
      }
    }
    __syncthreads();
  }

  // store Y as [b][s][h*128+dv] f32 (element: dv = 16j + r + mh, s = s0 + n)
#pragma unroll
  for (int j = 0; j < 8; ++j)
#pragma unroll
    for (int r = 0; r < 8; ++r) {
      const int dv = j * 16 + r + mh;
      const int s  = s0 + n;
      Y[((size_t)b * SEQ + s) * HID + h * HEAD + dv] = yacc[j][r];
    }
}

// ---------------- group-norm (per head) * gate -> f16 ----------------------
// grid: BATCH*SEQ  block: 256 (wave w == head w; 4 elems/lane)
__global__ __launch_bounds__(256)
void norm_gate_kernel(const float* __restrict__ Y, const _Float16* __restrict__ Gh,
                      const float* __restrict__ gw, const float* __restrict__ gb,
                      _Float16* __restrict__ Zh) {
  const int row  = blockIdx.x;                  // b*SEQ + s
  const int h    = threadIdx.x >> 5;
  const int lane = threadIdx.x & 31;
  const float* y = Y + (size_t)row * HID + h * HEAD;
  v4f v = *(const v4f*)(y + lane * 4);
  float s1 = v[0] + v[1] + v[2] + v[3];
  float s2 = v[0]*v[0] + v[1]*v[1] + v[2]*v[2] + v[3]*v[3];
#pragma unroll
  for (int off = 16; off > 0; off >>= 1) {
    s1 += __shfl_xor(s1, off, 32);
    s2 += __shfl_xor(s2, off, 32);
  }
  const float mu  = s1 * (1.0f / HEAD);
  const float var = s2 * (1.0f / HEAD) - mu * mu;
  const float inv = rsqrtf(var + GN_EPS);
#pragma unroll
  for (int i = 0; i < 4; ++i) {
    const int c = h * HEAD + lane * 4 + i;
    const float yn = (v[i] - mu) * inv * gw[c] + gb[c];
    const float g  = (float)Gh[(size_t)row * HID + c];
    Zh[(size_t)row * HID + c] = (_Float16)(g * yn);
  }
}

// ---------------------------------------------------------------------------

extern "C" void kernel_launch(void* const* d_in, const int* in_sizes, int n_in,
                              void* d_out, int out_size, void* d_ws, size_t ws_size,
                              hipStream_t stream) {
  const float* X    = (const float*)d_in[0];
  const float* W_Q  = (const float*)d_in[1];
  const float* W_K  = (const float*)d_in[2];
  const float* W_V  = (const float*)d_in[3];
  const float* W_G  = (const float*)d_in[4];
  const float* W_O  = (const float*)d_in[5];
  const float* gn_w = (const float*)d_in[6];
  const float* gn_b = (const float*)d_in[7];

  char* ws = (char*)d_ws;
  size_t off = 0;
  auto alloc = [&](size_t bytes) {
    void* p = ws + off;
    off += (bytes + 255) & ~(size_t)255;
    return p;
  };
  const size_t NX = (size_t)BATCH * SEQ * HID;
  _Float16* Xh  = (_Float16*)alloc(NX * 2);
  _Float16* Wqt = (_Float16*)alloc((size_t)NHEADS * HEAD * HID * 2);
  _Float16* Wkt = (_Float16*)alloc((size_t)NHEADS * HEAD * HID * 2);
  _Float16* Wvt = (_Float16*)alloc((size_t)NHEADS * HEAD * HID * 2);
  _Float16* Wgt = (_Float16*)alloc((size_t)HID * HID * 2);
  _Float16* Wot = (_Float16*)alloc((size_t)HID * HID * 2);
  _Float16* Qh  = (_Float16*)alloc(NX * 2);
  _Float16* Kh  = (_Float16*)alloc(NX * 2);
  _Float16* Vt  = (_Float16*)alloc(NX * 2);
  _Float16* Gh  = (_Float16*)alloc(NX * 2);
  float*    Yb  = (float*)alloc(NX * 4);
  _Float16* Zh  = (_Float16*)alloc(NX * 2);

  // 0) precision conversion + weight transposes (one-time, ~20 MB)
  f32_to_f16_kernel<<<(int)((NX + 255) / 256), 256, 0, stream>>>(X, Xh, (int)NX);
  transpose_to_f16_kernel<<<dim3(HEAD / 32, HID / 32, NHEADS), 256, 0, stream>>>(W_Q, Wqt, HID, HEAD);
  transpose_to_f16_kernel<<<dim3(HEAD / 32, HID / 32, NHEADS), 256, 0, stream>>>(W_K, Wkt, HID, HEAD);
  transpose_to_f16_kernel<<<dim3(HEAD / 32, HID / 32, NHEADS), 256, 0, stream>>>(W_V, Wvt, HID, HEAD);
  transpose_to_f16_kernel<<<dim3(HID / 32, HID / 32, 1), 256, 0, stream>>>(W_G, Wgt, HID, HID);
  transpose_to_f16_kernel<<<dim3(HID / 32, HID / 32, 1), 256, 0, stream>>>(W_O, Wot, HID, HID);

  // 1) QKV projection + xpos (Q,K) / transpose-store (V)
  proj_xpos_kernel<<<dim3(SEQ / 128, BATCH * NHEADS, 3), 256, 0, stream>>>(
      Xh, Wqt, Wkt, Wvt, Qh, Kh, Vt);

  // 2) gate = SiLU(X @ W_G)
  gemm_kernel<1><<<dim3(BATCH * SEQ / 128, HID / 128), 256, 0, stream>>>(
      Xh, Wgt, (void*)Gh, HID, HID);

  // 3) fused retention (flash-style, no SxS materialization)
  retention_kernel<<<dim3(SEQ / 128, BATCH * NHEADS), 256, 0, stream>>>(Qh, Kh, Vt, Yb);

  // 4) per-head group norm * gate
  norm_gate_kernel<<<BATCH * SEQ, 256, 0, stream>>>(Yb, Gh, gn_w, gn_b, Zh);

  // 5) out = Z @ W_O
  gemm_kernel<0><<<dim3(BATCH * SEQ / 128, HID / 128), 256, 0, stream>>>(
      Zh, Wot, d_out, HID, HID);
}
